// CAM_52003464020624
// MI455X (gfx1250) — matrix operand order repaired and checked by reading, not compiled, not run on hardware
//
#include <hip/hip_runtime.h>

// CAM channel-attention for MI455X (gfx1250, wave32).
// B=16, C=512, H=W=64 (HW=4096).
//   sim  = q @ q^T   (per batch)           -> bf16 WMMA, f32 accum
//   attn = softmax(rowmax - sim)           == exp(rowmin - sim)/sum
//   feat = attn @ q = attn @ (qT)^T        -> bf16 WMMA via pre-transposed qT
//   out  = gamma*feat + x
//
// qT is built once (streaming LDS transpose of x) and reuses qbf's workspace
// region (qbf is dead after GEMM1), so both GEMMs have purely contiguous
// 16-byte LDS fragment reads matching the ISA bf16 WMMA operand layouts.

typedef __attribute__((ext_vector_type(16))) __bf16 v16bf;
typedef __attribute__((ext_vector_type(8)))  float  v8f;

constexpr int BATCH = 16;
constexpr int CDIM  = 512;
constexpr int HW    = 4096;

constexpr int TILE = 128;       // output tile per 256-thread block
constexpr int KC   = 32;        // K chunk (matches wmma 16x16x32)
constexpr int LDA  = KC + 8;    // LDS row stride (bf16): 80B, conflict-avoiding

static __device__ __forceinline__ __bf16 to_bf16(float f) {
    unsigned u = __builtin_bit_cast(unsigned, f);
    u += 0x7FFFu + ((u >> 16) & 1u);           // round-to-nearest-even
    unsigned short h = (unsigned short)(u >> 16);
    return __builtin_bit_cast(__bf16, h);
}

union FragAB {
    uint4  u[2];
    __bf16 e[16];
    v16bf  v;
};

// ---------------------------------------------------------------- kernel 1
// f32 -> bf16 cast of q (= x reshaped). 4 elements / thread.
__global__ __launch_bounds__(256) void cast_bf16_kernel(
    const float* __restrict__ x, __bf16* __restrict__ qbf)
{
    const size_t i4   = (size_t)blockIdx.x * 256 + threadIdx.x;
    const size_t base = i4 * 4;
    float4 f = *(const float4*)(x + base);
    union { __bf16 e[4]; uint2 u; } p;
    p.e[0] = to_bf16(f.x); p.e[1] = to_bf16(f.y);
    p.e[2] = to_bf16(f.z); p.e[3] = to_bf16(f.w);
    *(uint2*)(qbf + base) = p.u;
}

// ---------------------------------------------------------------- kernel 2
// sim[b] = q[b] @ q[b]^T.  Block: 128x128 tile, 8 waves of 32x64.
__global__ __launch_bounds__(256) void sim_gemm_kernel(
    const __bf16* __restrict__ q, float* __restrict__ sim)
{
    __shared__ __align__(16) __bf16 As[TILE][LDA];
    __shared__ __align__(16) __bf16 Bs[TILE][LDA];

    const int b     = blockIdx.z;
    const int dbase = blockIdx.x * TILE;   // sim columns (rows of q as K^T)
    const int cbase = blockIdx.y * TILE;   // sim rows
    const __bf16* qb = q + (size_t)b * CDIM * HW;

    const int tid  = threadIdx.x;
    const int lane = tid & 31;
    const int w    = tid >> 5;          // 0..7
    const int wrow = w >> 1;            // 0..3  -> 32-row stripe
    const int wcol = w & 1;             // 0..1  -> 64-col stripe

    const int frow = tid >> 1;          // LDS fill: row 0..127
    const int fcol = (tid & 1) * 16;    // 16 bf16 (32B) halves

    v8f acc[2][4] = {};

    for (int k0 = 0; k0 < HW; k0 += KC) {
        const __bf16* ga = qb + (size_t)(cbase + frow) * HW + k0 + fcol;
        const __bf16* gb = qb + (size_t)(dbase + frow) * HW + k0 + fcol;
        uint4 a0 = *(const uint4*)(ga);
        uint4 a1 = *(const uint4*)(ga + 8);
        uint4 b0 = *(const uint4*)(gb);
        uint4 b1 = *(const uint4*)(gb + 8);
        __builtin_prefetch(ga + KC, 0, 1);   // global_prefetch_b8: next K chunk
        __builtin_prefetch(gb + KC, 0, 1);
        *(uint4*)&As[frow][fcol]     = a0;
        *(uint4*)&As[frow][fcol + 8] = a1;
        *(uint4*)&Bs[frow][fcol]     = b0;
        *(uint4*)&Bs[frow][fcol + 8] = b1;
        __syncthreads();

        // A frag (ISA 16-bit 16x32 layout): lane m=l&15, K {0..7,16..23}+8*(l>=16)
        const int kg = (lane >> 4) << 3;     // 0 | 8
        const int ml = lane & 15;
        FragAB afr[2];
#pragma unroll
        for (int mt = 0; mt < 2; ++mt) {
            const int m = wrow * 32 + mt * 16 + ml;
            afr[mt].u[0] = *(const uint4*)&As[m][kg];
            afr[mt].u[1] = *(const uint4*)&As[m][kg + 16];
        }
        // B frag (32x16): lane n=l&15, K {0..7,16..23}+8*(l>=16) (same as A)
        const int kq = (lane >> 4) << 4;     // 0 | 16
        FragAB bfr[4];
#pragma unroll
        for (int nt = 0; nt < 4; ++nt) {
            const int d = wcol * 64 + nt * 16 + ml;
            bfr[nt].u[0] = *(const uint4*)&Bs[d][kg];
            bfr[nt].u[1] = *(const uint4*)&Bs[d][kg + 16];
        }
#pragma unroll
        for (int mt = 0; mt < 2; ++mt)
#pragma unroll
            for (int nt = 0; nt < 4; ++nt)
                acc[mt][nt] = __builtin_amdgcn_wmma_f32_16x16x32_bf16(
                    false, afr[mt].v, false, bfr[nt].v,
                    (short)0, acc[mt][nt], false, false);
        __syncthreads();
    }

    // D layout: VGPR i: lanes 0-15 -> M=i, lanes 16-31 -> M=i+8; N = lane&15
    float* simb = sim + (size_t)b * CDIM * CDIM;
    const int mhi = (lane >> 4) << 3;
    const int nl  = lane & 15;
#pragma unroll
    for (int mt = 0; mt < 2; ++mt)
#pragma unroll
        for (int nt = 0; nt < 4; ++nt)
#pragma unroll
            for (int i = 0; i < 8; ++i) {
                const int c = cbase + wrow * 32 + mt * 16 + i + mhi;
                const int d = dbase + wcol * 64 + nt * 16 + nl;
                simb[(size_t)c * CDIM + d] = acc[mt][nt][i];
            }
}

// ---------------------------------------------------------------- kernel 3
// softmax(rowmax - sim) == exp(rowmin - sim) / sum(exp(rowmin - sim))
__global__ __launch_bounds__(256) void softmax_kernel(
    const float* __restrict__ sim, __bf16* __restrict__ attn)
{
    __shared__ float red[256];
    const size_t row  = blockIdx.x;            // B*C rows
    const float* base = sim + row * CDIM;
    const int tid = threadIdx.x;

    const float v0 = base[tid];
    const float v1 = base[tid + 256];

    red[tid] = fminf(v0, v1);
    __syncthreads();
#pragma unroll
    for (int s = 128; s > 0; s >>= 1) {
        if (tid < s) red[tid] = fminf(red[tid], red[tid + s]);
        __syncthreads();
    }
    const float mn = red[0];
    __syncthreads();

    const float e0 = __expf(mn - v0);
    const float e1 = __expf(mn - v1);
    red[tid] = e0 + e1;
    __syncthreads();
#pragma unroll
    for (int s = 128; s > 0; s >>= 1) {
        if (tid < s) red[tid] += red[tid + s];
        __syncthreads();
    }
    const float inv = 1.0f / red[0];

    __bf16* arow = attn + row * CDIM;
    arow[tid]       = to_bf16(e0 * inv);
    arow[tid + 256] = to_bf16(e1 * inv);
}

// ---------------------------------------------------------------- kernel 4
// qT[b][hw][c] = bf16(x[b][c][hw]).  64x64 LDS tile transpose, one-shot pass.
// Written over qbf's region (qbf is dead after GEMM1); source is original x.
__global__ __launch_bounds__(256) void transpose_cast_kernel(
    const float* __restrict__ x, __bf16* __restrict__ qT)
{
    __shared__ __align__(16) __bf16 T[64][72];   // 144B row stride
    const int b  = blockIdx.z;
    const int hb = blockIdx.x * 64;   // hw tile base
    const int cb = blockIdx.y * 64;   // channel tile base
    const float* xb  = x  + (size_t)b * CDIM * HW;
    __bf16*      qTb = qT + (size_t)b * HW * CDIM;

    const int t   = threadIdx.x;
    const int row = t >> 2;           // 0..63
    const int col = (t & 3) * 16;     // 0,16,32,48

    // phase 1: coalesced read of x rows (c-major), cast, stage tile in LDS
    const float* src = xb + (size_t)(cb + row) * HW + hb + col;
    union { __bf16 e[16]; uint4 u[2]; } pk;
#pragma unroll
    for (int j = 0; j < 16; j += 4) {
        float4 f = *(const float4*)(src + j);
        pk.e[j + 0] = to_bf16(f.x); pk.e[j + 1] = to_bf16(f.y);
        pk.e[j + 2] = to_bf16(f.z); pk.e[j + 3] = to_bf16(f.w);
    }
    *(uint4*)&T[row][col]     = pk.u[0];
    *(uint4*)&T[row][col + 8] = pk.u[1];
    __syncthreads();

    // phase 2: gather tile columns, coalesced write of qT rows (hw-major)
    union { __bf16 e[16]; uint4 u[2]; } o;
#pragma unroll
    for (int j = 0; j < 16; ++j) o.e[j] = T[col + j][row];
    __bf16* dst = qTb + (size_t)(hb + row) * CDIM + cb + col;
    *(uint4*)(dst)     = o.u[0];
    *(uint4*)(dst + 8) = o.u[1];
}

// ---------------------------------------------------------------- kernel 5
// out = gamma * (attn @ qT^T) + x.  Structurally identical to GEMM1:
// A = attn rows (c, K=d contiguous), B = qT rows (n, K=d contiguous).
__global__ __launch_bounds__(256) void feat_gemm_kernel(
    const __bf16* __restrict__ attn, const __bf16* __restrict__ qT,
    const float* __restrict__ x, const float* __restrict__ gamma,
    float* __restrict__ out)
{
    __shared__ __align__(16) __bf16 As[TILE][LDA];   // attn tile: [c][d]
    __shared__ __align__(16) __bf16 Bs[TILE][LDA];   // qT tile:   [n][d]

    const int b     = blockIdx.z;
    const int nbase = blockIdx.x * TILE;   // hw columns
    const int cbase = blockIdx.y * TILE;   // channel rows
    const __bf16* ab  = attn + (size_t)b * CDIM * CDIM;
    const __bf16* qTb = qT   + (size_t)b * HW * CDIM;

    const int tid  = threadIdx.x;
    const int lane = tid & 31;
    const int w    = tid >> 5;
    const int wrow = w >> 1;
    const int wcol = w & 1;

    const int frow = tid >> 1;
    const int fcol = (tid & 1) * 16;

    const float g = gamma[0];
    v8f acc[2][4] = {};

    for (int k0 = 0; k0 < CDIM; k0 += KC) {
        const __bf16* ga = ab  + (size_t)(cbase + frow) * CDIM + k0 + fcol;
        const __bf16* gb = qTb + (size_t)(nbase + frow) * CDIM + k0 + fcol;
        uint4 a0 = *(const uint4*)(ga);
        uint4 a1 = *(const uint4*)(ga + 8);
        uint4 b0 = *(const uint4*)(gb);
        uint4 b1 = *(const uint4*)(gb + 8);
        __builtin_prefetch(ga + KC, 0, 1);
        __builtin_prefetch(gb + KC, 0, 1);
        *(uint4*)&As[frow][fcol]     = a0;
        *(uint4*)&As[frow][fcol + 8] = a1;
        *(uint4*)&Bs[frow][fcol]     = b0;
        *(uint4*)&Bs[frow][fcol + 8] = b1;
        __syncthreads();

        const int kg = (lane >> 4) << 3;
        const int ml = lane & 15;
        FragAB afr[2];
#pragma unroll
        for (int mt = 0; mt < 2; ++mt) {
            const int m = wrow * 32 + mt * 16 + ml;
            afr[mt].u[0] = *(const uint4*)&As[m][kg];
            afr[mt].u[1] = *(const uint4*)&As[m][kg + 16];
        }
        const int kq = (lane >> 4) << 4;
        FragAB bfr[4];
#pragma unroll
        for (int nt = 0; nt < 4; ++nt) {
            const int n = wcol * 64 + nt * 16 + ml;
            bfr[nt].u[0] = *(const uint4*)&Bs[n][kg];
            bfr[nt].u[1] = *(const uint4*)&Bs[n][kg + 16];
        }
#pragma unroll
        for (int mt = 0; mt < 2; ++mt)
#pragma unroll
            for (int nt = 0; nt < 4; ++nt)
                acc[mt][nt] = __builtin_amdgcn_wmma_f32_16x16x32_bf16(
                    false, afr[mt].v, false, bfr[nt].v,
                    (short)0, acc[mt][nt], false, false);
        __syncthreads();
    }

    const float* xb = x   + (size_t)b * CDIM * HW;
    float*       ob = out + (size_t)b * CDIM * HW;
    const int mhi = (lane >> 4) << 3;
    const int nl  = lane & 15;
#pragma unroll
    for (int mt = 0; mt < 2; ++mt)
#pragma unroll
        for (int nt = 0; nt < 4; ++nt)
#pragma unroll
            for (int i = 0; i < 8; ++i) {
                const int c = cbase + wrow * 32 + mt * 16 + i + mhi;
                const int n = nbase + wcol * 64 + nt * 16 + nl;
                const size_t idx = (size_t)c * HW + n;
                ob[idx] = g * acc[mt][nt][i] + xb[idx];
            }
}

// ---------------------------------------------------------------- launch
extern "C" void kernel_launch(void* const* d_in, const int* in_sizes, int n_in,
                              void* d_out, int out_size, void* d_ws, size_t ws_size,
                              hipStream_t stream) {
    const float* x     = (const float*)d_in[0];
    const float* gamma = (const float*)d_in[1];
    float*       out   = (float*)d_out;

    // workspace layout (peak ~92.3 MB):
    //   [0, 64MiB)        qbf (row-major bf16)  -> later reused for qT
    //   [64MiB, 80MiB)    sim (f32)
    //   [80MiB, 88MiB)    attn (bf16)
    char* ws = (char*)d_ws;
    const size_t qbf_bytes = (size_t)BATCH * CDIM * HW * 2;        // 67,108,864
    const size_t sim_bytes = (size_t)BATCH * CDIM * CDIM * 4;      // 16,777,216
    __bf16* qbf  = (__bf16*)ws;
    float*  sim  = (float*)(ws + qbf_bytes);
    __bf16* attn = (__bf16*)(ws + qbf_bytes + sim_bytes);
    __bf16* qT   = qbf;   // overwrites qbf after it is dead

    const int total = BATCH * CDIM * HW;                            // 33,554,432
    cast_bf16_kernel<<<dim3(total / (256 * 4)), 256, 0, stream>>>(x, qbf);

    sim_gemm_kernel<<<dim3(CDIM / TILE, CDIM / TILE, BATCH), 256, 0, stream>>>(qbf, sim);

    softmax_kernel<<<dim3(BATCH * CDIM), 256, 0, stream>>>(sim, attn);

    // qbf dead from here; build qT over its region from the original f32 x
    transpose_cast_kernel<<<dim3(HW / 64, CDIM / 64, BATCH), 256, 0, stream>>>(x, qT);

    feat_gemm_kernel<<<dim3(HW / TILE, CDIM / TILE, BATCH), 256, 0, stream>>>(
        attn, qT, x, gamma, out);
}
